// GCNLayer_24893630448148
// MI455X (gfx1250) — matrix operand order repaired and checked by reading, not compiled, not run on hardware
//
#include <hip/hip_runtime.h>
#include <hip/hip_bf16.h>

// Problem constants (match reference)
#define BB   32
#define SS   256
#define DD   256
#define UU   256
#define LL   16
#define DEG  16
#define NN   (BB * SS)      // 8192
#define EE   (NN * DEG)     // 131072

typedef _Float16 v16h __attribute__((ext_vector_type(16)));
typedef _Float16 h8   __attribute__((ext_vector_type(8)));
typedef float    v8f  __attribute__((ext_vector_type(8)));

// ---------------------------------------------------------------------------
// 1) Convert x (= transpose of src [S,B,D] -> [N=B*S, D]) to f16
// ---------------------------------------------------------------------------
__global__ __launch_bounds__(256) void prep_x_kernel(const float* __restrict__ src,
                                                     _Float16* __restrict__ xh) {
    int i = blockIdx.x * 256 + threadIdx.x;      // over N*D
    int n = i >> 8, d = i & 255;
    int b = n >> 8, s = n & 255;                 // n = b*S + s
    xh[i] = (_Float16)src[(s * BB + b) * DD + d];
}

// ---------------------------------------------------------------------------
// 2) Transpose + convert weights: V[L,D,U] -> Vt[L,U,D] (f16); W_self likewise
// ---------------------------------------------------------------------------
__global__ __launch_bounds__(256) void prep_w_kernel(const float* __restrict__ V_in,
                                                     const float* __restrict__ V_out,
                                                     const float* __restrict__ W_self,
                                                     _Float16* __restrict__ Vt_in,
                                                     _Float16* __restrict__ Vt_out,
                                                     _Float16* __restrict__ Wt_self) {
    const int LDU = LL * DD * UU;                // 1,048,576
    int i = blockIdx.x * 256 + threadIdx.x;
    if (i < LDU) {
        int l = i >> 16, r = i & 65535, d = r >> 8, u = r & 255;
        int o = (l << 16) + (u << 8) + d;
        Vt_in[o]  = (_Float16)V_in[i];
        Vt_out[o] = (_Float16)V_out[i];
    } else {
        int j = i - LDU;
        if (j < DD * UU) {
            int d = j >> 8, u = j & 255;
            Wt_self[u * DD + d] = (_Float16)W_self[j];
        }
    }
}

// ---------------------------------------------------------------------------
// 3) Gate dot products: one wave (32 lanes) per node, three dots of length D
// ---------------------------------------------------------------------------
__global__ __launch_bounds__(256) void gates_kernel(const float* __restrict__ src,
                                                    const float* __restrict__ Vg_in,
                                                    const float* __restrict__ Vg_out,
                                                    const float* __restrict__ Wg_self,
                                                    float* __restrict__ gin,
                                                    float* __restrict__ gout,
                                                    float* __restrict__ gself) {
    int wave = threadIdx.x >> 5, lane = threadIdx.x & 31;
    int n = blockIdx.x * 8 + wave;
    int b = n >> 8, s = n & 255;
    const float* xrow = src + (s * BB + b) * DD;
    float a0 = 0.f, a1 = 0.f, a2 = 0.f;
    for (int d = lane; d < DD; d += 32) {
        float xv = xrow[d];
        a0 += xv * Vg_in[d];
        a1 += xv * Vg_out[d];
        a2 += xv * Wg_self[d];
    }
    for (int off = 16; off > 0; off >>= 1) {
        a0 += __shfl_down(a0, off, 32);
        a1 += __shfl_down(a1, off, 32);
        a2 += __shfl_down(a2, off, 32);
    }
    if (lane == 0) { gin[n] = a0; gout[n] = a1; gself[n] = a2; }
}

// ---------------------------------------------------------------------------
// 4) Dual GEMM: Y0 = xh @ Bt0^T, Y1 = xh @ Bt1^T (Bt stored [U,D] f16).
//    Block = 8 waves; waves 0-3 -> Bt0/Y0, waves 4-7 -> Bt1/Y1.
//    Each wave: 4 WMMA 16x16 C tiles (64 cols); A tile (16 rows x 256 K) in LDS.
// ---------------------------------------------------------------------------
__global__ __launch_bounds__(256) void gemm_dual_kernel(const _Float16* __restrict__ xh,
                                                        const _Float16* __restrict__ Bt0,
                                                        const _Float16* __restrict__ Bt1,
                                                        float* __restrict__ Y0,
                                                        float* __restrict__ Y1) {
    __shared__ _Float16 As[16 * DD];             // 8 KB
    const int tid = threadIdx.x;
    const int row0 = blockIdx.x * 16;

    // Stage A: 4096 halves, 16 halves (32B) per thread
    {
        int t = tid * 16;
        int r = t >> 8, c = t & 255;
        *(v16h*)(As + t) = *(const v16h*)(xh + (size_t)(row0 + r) * DD + c);
    }
    __syncthreads();

    const int wave = tid >> 5;
    const int lane = tid & 31;
    const _Float16* Bt = (wave < 4) ? Bt0 : Bt1;
    float* Y = (wave < 4) ? Y0 : Y1;
    if (Bt == nullptr) return;

    const int colBase = (wave & 3) * 64;
    const int m   = lane & 15;
    const int kba = (lane < 16) ? 0 : 16;        // A fragment K sub-offset
    const int nl  = lane & 15;
    const int kbb = (lane < 16) ? 0 : 16;        // B fragment K sub-offset

    v8f c0 = {}, c1 = {}, c2 = {}, c3 = {};
    for (int k0 = 0; k0 < DD; k0 += 32) {
        // A fragment: halves 0..7 = A[m][k0+kba .. +7], 8..15 = A[m][k0+kba+8 .. +15]
        h8 lo = *(const h8*)(As + m * DD + k0 + kba);
        h8 hi = *(const h8*)(As + m * DD + k0 + kba + 8);
        v16h a = __builtin_shufflevector(lo, hi, 0, 1, 2, 3, 4, 5, 6, 7,
                                                 8, 9, 10, 11, 12, 13, 14, 15);
        // B fragments: lane holds column n, 16 contiguous K halves (Bt is [U,D])
        v16h b0 = *(const v16h*)(Bt + (size_t)(colBase + 0 * 16 + nl) * DD + k0 + kbb);
        v16h b1 = *(const v16h*)(Bt + (size_t)(colBase + 1 * 16 + nl) * DD + k0 + kbb);
        v16h b2 = *(const v16h*)(Bt + (size_t)(colBase + 2 * 16 + nl) * DD + k0 + kbb);
        v16h b3 = *(const v16h*)(Bt + (size_t)(colBase + 3 * 16 + nl) * DD + k0 + kbb);
        c0 = __builtin_amdgcn_wmma_f32_16x16x32_f16(false, a, false, b0, (short)0, c0, false, false);
        c1 = __builtin_amdgcn_wmma_f32_16x16x32_f16(false, a, false, b1, (short)0, c1, false, false);
        c2 = __builtin_amdgcn_wmma_f32_16x16x32_f16(false, a, false, b2, (short)0, c2, false, false);
        c3 = __builtin_amdgcn_wmma_f32_16x16x32_f16(false, a, false, b3, (short)0, c3, false, false);
    }

    // C layout: VGPR r -> row m = r + 8*(lane>=16); col = lane&15
    const int mrow = (lane >> 4) * 8;
    #pragma unroll
    for (int r = 0; r < 8; ++r) {
        size_t base = (size_t)(row0 + mrow + r) * UU;
        Y[base + colBase + 0 * 16 + nl] = c0[r];
        Y[base + colBase + 1 * 16 + nl] = c1[r];
        Y[base + colBase + 2 * 16 + nl] = c2[r];
        Y[base + colBase + 3 * 16 + nl] = c3[r];
    }
}

// ---------------------------------------------------------------------------
// 5) Initialize accumulator with self-loop branch
// ---------------------------------------------------------------------------
__global__ __launch_bounds__(256) void init_acc_kernel(const float* __restrict__ Ysame,
                                                       const float* __restrict__ gself,
                                                       const float* __restrict__ mask_loop,
                                                       float* __restrict__ acc) {
    int i = blockIdx.x * 256 + threadIdx.x;      // over N*U
    int n = i >> 8;
    float p = (1.0f / (1.0f + __expf(-gself[n]))) * mask_loop[n];
    acc[i] = p * Ysame[i];
}

// ---------------------------------------------------------------------------
// 6) Deterministic gather-accumulate for one label (both branches).
//    Block = one output node; thread = one U column; exclusive acc ownership.
// ---------------------------------------------------------------------------
__global__ __launch_bounds__(256) void gather_label_kernel(const float* __restrict__ Yin,
                                                           const float* __restrict__ Yout,
                                                           const int* __restrict__ arc_in,
                                                           const int* __restrict__ arc_out,
                                                           const int* __restrict__ lab_in,
                                                           const int* __restrict__ lab_out,
                                                           const float* __restrict__ b_in,
                                                           const float* __restrict__ b_out,
                                                           const float* __restrict__ bg_in,
                                                           const float* __restrict__ bg_out,
                                                           const float* __restrict__ gin,
                                                           const float* __restrict__ gout,
                                                           const float* __restrict__ mask_in,
                                                           const float* __restrict__ mask_out,
                                                           float* __restrict__ acc,
                                                           int label) {
    const int n = blockIdx.x;
    const int u = threadIdx.x;

    // Scalar (uniform) early-out: does this node touch `label` at all?
    bool any = false;
    #pragma unroll
    for (int d = 0; d < DEG; ++d) {
        int e = n * DEG + d;
        any = any || (lab_in[e] == label) || (lab_out[e] == label);
    }
    if (!any) return;

    float s = 0.f;
    const float bin_u  = b_in [label * UU + u];
    const float bout_u = b_out[label * UU + u];
    const float bgin   = bg_in [label];
    const float bgout  = bg_out[label];

    #pragma unroll
    for (int d = 0; d < DEG; ++d) {
        int e = n * DEG + d;
        if (lab_in[e] == label) {
            int src = arc_in[e] * SS + arc_in[EE + e];
            float g = gin[src] + bgin;
            float w = (1.0f / (1.0f + __expf(-g))) * mask_in[e];
            s += w * (Yin[(size_t)src * UU + u] + bin_u);
        }
        if (lab_out[e] == label) {
            int src = arc_out[e] * SS + arc_out[EE + e];
            float g = gout[src] + bgout;
            float w = (1.0f / (1.0f + __expf(-g))) * mask_out[e];
            s += w * (Yout[(size_t)src * UU + u] + bout_u);
        }
    }
    acc[(size_t)n * UU + u] += s;
}

// ---------------------------------------------------------------------------
// 7) ReLU, sent_mask, transpose [N,U] -> [S,B,U]
// ---------------------------------------------------------------------------
__global__ __launch_bounds__(256) void finish_kernel(const float* __restrict__ acc,
                                                     const float* __restrict__ sent_mask,
                                                     float* __restrict__ out) {
    int i = blockIdx.x * 256 + threadIdx.x;      // over N*U
    int n = i >> 8, u = i & 255;
    int b = n >> 8, s = n & 255;
    float v = acc[i];
    v = (v > 0.f) ? v : 0.f;
    out[(size_t)(s * BB + b) * UU + u] = v * sent_mask[s * BB + b];
}

// ---------------------------------------------------------------------------
extern "C" void kernel_launch(void* const* d_in, const int* in_sizes, int n_in,
                              void* d_out, int out_size, void* d_ws, size_t ws_size,
                              hipStream_t stream) {
    const float* src        = (const float*)d_in[0];
    const float* V_in       = (const float*)d_in[1];
    const float* b_in       = (const float*)d_in[2];
    const float* Vg_in      = (const float*)d_in[3];
    const float* bg_in      = (const float*)d_in[4];
    const float* V_out      = (const float*)d_in[5];
    const float* b_out      = (const float*)d_in[6];
    const float* Vg_out     = (const float*)d_in[7];
    const float* bg_out     = (const float*)d_in[8];
    const float* W_self     = (const float*)d_in[9];
    const float* Wg_self    = (const float*)d_in[10];
    const int*   arc_in     = (const int*)d_in[11];
    const int*   arc_out    = (const int*)d_in[12];
    const int*   lab_in     = (const int*)d_in[13];
    const int*   lab_out    = (const int*)d_in[14];
    const float* mask_in    = (const float*)d_in[15];
    const float* mask_out   = (const float*)d_in[16];
    const float* mask_loop  = (const float*)d_in[17];
    const float* sent_mask  = (const float*)d_in[18];
    float* out = (float*)d_out;

    // Workspace carve-out (256B aligned slices)
    size_t off = 0;
    char* base = (char*)d_ws;
    auto carve = [&](size_t bytes) -> char* {
        char* p = base + off;
        off += (bytes + 255) & ~(size_t)255;
        return p;
    };
    _Float16* xh     = (_Float16*)carve((size_t)NN * DD * 2);       // 4 MB
    _Float16* Vt_in  = (_Float16*)carve((size_t)LL * DD * UU * 2);  // 2 MB
    _Float16* Vt_out = (_Float16*)carve((size_t)LL * DD * UU * 2);  // 2 MB
    _Float16* Wt     = (_Float16*)carve((size_t)DD * UU * 2);       // 128 KB
    float* gin   = (float*)carve((size_t)NN * 4);
    float* gout  = (float*)carve((size_t)NN * 4);
    float* gself = (float*)carve((size_t)NN * 4);
    float* Yin   = (float*)carve((size_t)NN * UU * 4);              // 8 MB
    float* Yout  = (float*)carve((size_t)NN * UU * 4);              // 8 MB
    float* acc   = (float*)carve((size_t)NN * UU * 4);              // 8 MB

    // 1) x -> f16 (with transpose)
    prep_x_kernel<<<(NN * DD) / 256, 256, 0, stream>>>(src, xh);
    // 2) weights -> transposed f16
    {
        int total = LL * DD * UU + DD * UU;
        prep_w_kernel<<<(total + 255) / 256, 256, 0, stream>>>(V_in, V_out, W_self,
                                                               Vt_in, Vt_out, Wt);
    }
    // 3) gate dots
    gates_kernel<<<NN / 8, 256, 0, stream>>>(src, Vg_in, Vg_out, Wg_self, gin, gout, gself);

    // 4) self-loop GEMM (waves 4-7 idle) + init accumulator
    gemm_dual_kernel<<<NN / 16, 256, 0, stream>>>(xh, Wt, nullptr, Yin, nullptr);
    init_acc_kernel<<<(NN * UU) / 256, 256, 0, stream>>>(Yin, gself, mask_loop, acc);

    // 5) per-label dual GEMM + deterministic gather
    for (int l = 0; l < LL; ++l) {
        const _Float16* Bl_in  = Vt_in  + (size_t)l * DD * UU;
        const _Float16* Bl_out = Vt_out + (size_t)l * DD * UU;
        gemm_dual_kernel<<<NN / 16, 256, 0, stream>>>(xh, Bl_in, Bl_out, Yin, Yout);
        gather_label_kernel<<<NN, 256, 0, stream>>>(Yin, Yout, arc_in, arc_out,
                                                    lab_in, lab_out, b_in, b_out,
                                                    bg_in, bg_out, gin, gout,
                                                    mask_in, mask_out, acc, l);
    }

    // 6) relu + sent_mask + transpose to [S,B,U]
    finish_kernel<<<(NN * UU) / 256, 256, 0, stream>>>(acc, sent_mask, out);
}
